// WindowTimeToTokenAttention_15547781612055
// MI455X (gfx1250) — matrix operand order repaired and checked by reading, not compiled, not run on hardware
//
#include <hip/hip_runtime.h>

// ---------------------------------------------------------------------------
// WindowTimeToTokenAttention for MI455X (gfx1250, wave32, WMMA bf16).
//
// Reference dead code: v = kv_in @ v_w.T is never consumed (attn uses the
// projected q as values) -> v_w/v_b skipped. First NQ rows of each group's
// output are a raw copy of x.
//
// ~640 GFLOP (three 1024-deep projections dominate) vs ~1.3 GB HBM floor
// (~56us @ 23.3 TB/s): run all matmuls on v_wmma_f32_16x16x32_bf16 (f32 acc).
// CDNA5 data movement: weights pre-converted to bf16 once, then GEMM tiles
// are staged with GLOBAL_LOAD_ASYNC_TO_LDS_B128 (ASYNCcnt) into a
// double-buffered LDS pipeline so global latency hides under the WMMAs.
// Scratch: Qbuf bf16 (20MB) + Kbuf bf16 (302MB) + 3x bf16 weights (6MB);
// attention output overwrites Kbuf in place.
// ---------------------------------------------------------------------------

typedef __attribute__((ext_vector_type(16))) __bf16 v16bf;
typedef __attribute__((ext_vector_type(8)))  __bf16 v8bf;
typedef __attribute__((ext_vector_type(8)))  float  v8f;

constexpr int E_DIM   = 1024;  // embedding
constexpr int GROUPSZ = 154;
constexpr int NQ_     = 10;
constexpr int NKV_    = 144;
constexpr int NH      = 16;    // heads
constexpr int HD      = 64;    // head dim
constexpr int NGROUPS = 1024;

static __device__ __forceinline__ v16bf cat16(v8bf lo, v8bf hi) {
  return __builtin_shufflevector(lo, hi, 0,1,2,3,4,5,6,7,8,9,10,11,12,13,14,15);
}

static __device__ __forceinline__ v8f wmma_bf16(v16bf a, v16bf b, v8f c) {
  // D = A(16x32 bf16) * B(32x16 bf16) + C(16x16 f32)
  return __builtin_amdgcn_wmma_f32_16x16x32_bf16(
      /*neg_a=*/false, a, /*neg_b=*/false, b,
      /*c_mod=*/(short)0, c, /*reuse_a=*/false, /*reuse_b=*/false);
}

static __device__ __forceinline__ v8bf cvt8(const float* __restrict__ s) {
  const float4 f0 = *(const float4*)(s);
  const float4 f1 = *(const float4*)(s + 4);
  v8bf r = { (__bf16)f0.x, (__bf16)f0.y, (__bf16)f0.z, (__bf16)f0.w,
             (__bf16)f1.x, (__bf16)f1.y, (__bf16)f1.z, (__bf16)f1.w };
  return r;
}

// Async global->LDS copy (16B per lane).  The LDS aperture occupies the upper
// 32 bits of a flat shared address (ISA 10.2), so the
// low 32 bits of a __shared__ pointer are exactly the dsaddr the instruction
// needs.  Tracked by ASYNCcnt.
static __device__ __forceinline__ void async_cp_b128(void* lds_dst, const void* gsrc) {
  const unsigned d = (unsigned)(uintptr_t)lds_dst;
  asm volatile("global_load_async_to_lds_b128 %0, %1, off"
               :: "v"(d), "v"(gsrc) : "memory");
}
static __device__ __forceinline__ void wait_async0() {
  asm volatile("s_wait_asynccnt 0x0" ::: "memory");
}

// ---------------------------------------------------------------------------
// Kernel 0: raw copy of q_in rows (first NQ rows of each group) into d_out.
// ---------------------------------------------------------------------------
__global__ __launch_bounds__(256)
void copy_qin(const float* __restrict__ x, float* __restrict__ y) {
  const long r   = blockIdx.x;               // 0 .. G*NQ-1
  const long g   = r / NQ_;
  const long row = g * GROUPSZ + (r - g * NQ_);
  const float4* src = (const float4*)(x + row * E_DIM);
  float4*       dst = (float4*)(y + row * E_DIM);
  dst[threadIdx.x] = src[threadIdx.x];
}

// ---------------------------------------------------------------------------
// Kernel 1: one-shot f32 -> bf16 weight conversion (W is 1024x1024).
// 512 blocks x 256 threads x 8 elements.
// ---------------------------------------------------------------------------
__global__ __launch_bounds__(256)
void cvt_w(const float* __restrict__ w, __bf16* __restrict__ wb) {
  const long i = (((long)blockIdx.x << 8) + threadIdx.x) << 3;
  *(v8bf*)(wb + i) = cvt8(w + i);
}

// ---------------------------------------------------------------------------
// Tiled GEMM: C[M x 1024] = A[M x 1024] @ W^T + bias,  W pre-converted bf16.
//   OPROJ == false : A = x (f32) with per-group row gather (MG rows/group at
//                    offset OFF); A staged via load+cvt; output bf16 linear.
//   OPROJ == true  : A = bf16 scratch (linear rows); A staged via async LDS
//                    copy; output f32 scattered into d_out.
// Block: 256 threads (8 waves), tile 128x128, K-step 32, double-buffered LDS.
// Wave (mw 0..1, nw 0..3) owns a 64x32 sub-tile = 4x2 WMMA accumulators.
// ---------------------------------------------------------------------------
template<bool OPROJ, int MG, int OFF>
__global__ __launch_bounds__(256)
void proj_gemm(const float*  __restrict__ Af,
               const __bf16* __restrict__ Ab,
               const __bf16* __restrict__ Wb,
               const float*  __restrict__ bias,
               __bf16* __restrict__ OutB,
               float*  __restrict__ OutF)
{
  constexpr int LDA    = 40;              // 32 bf16 + 8 pad (bank stagger)
  constexpr int KITERS = E_DIM / 32;      // 32
  __shared__ __bf16 As[2][128 * LDA];
  __shared__ __bf16 Bs[2][128 * LDA];     // Bs[n][k] = Wb[n0+n][k0+k]

  const int tid  = threadIdx.x;
  const int lane = tid & 31;
  const int wave = tid >> 5;
  const int mw   = wave >> 2;             // 0..1
  const int nw   = wave & 3;              // 0..3
  const int l15  = lane & 15;
  const int lh   = lane >> 4;             // lane half selects K sub-range

  const long m0 = (long)blockIdx.y * 128;
  const long n0 = (long)blockIdx.x * 128;

  v8f acc[4][2];
#pragma unroll
  for (int i = 0; i < 4; ++i)
#pragma unroll
    for (int j = 0; j < 2; ++j)
      acc[i][j] = (v8f){0.f,0.f,0.f,0.f,0.f,0.f,0.f,0.f};

  // Staging assignment: thread -> (row, 16-elem half-row).
  const int srow = tid >> 1;
  const int scol = (tid & 1) << 4;

  long a_src_row;
  if constexpr (!OPROJ) {
    const long r = m0 + srow;
    const long g = r / MG;                         // MG is compile-time
    a_src_row = g * GROUPSZ + OFF + (r - g * MG);
  } else {
    a_src_row = m0 + srow;
  }
  const long w_src_row = n0 + srow;

  auto stage = [&](int k0, int buf) {
    // B tile: pure bf16 bytes -> async global->LDS copy.
    const __bf16* bsrc = Wb + w_src_row * E_DIM + k0 + scol;
    __bf16* bdst = &Bs[buf][srow * LDA + scol];
    async_cp_b128(bdst,     bsrc);
    async_cp_b128(bdst + 8, bsrc + 8);
    __bf16* adst = &As[buf][srow * LDA + scol];
    if constexpr (OPROJ) {
      const __bf16* asrc = Ab + a_src_row * E_DIM + k0 + scol;
      async_cp_b128(adst,     asrc);
      async_cp_b128(adst + 8, asrc + 8);
    } else {
      const float* asrc = Af + a_src_row * E_DIM + k0 + scol;
      *(v8bf*)adst       = cvt8(asrc);
      *(v8bf*)(adst + 8) = cvt8(asrc + 8);
    }
  };

  stage(0, 0);
  wait_async0();
  __syncthreads();

  for (int it = 0; it < KITERS; ++it) {
    const int buf = it & 1;
    if (it + 1 < KITERS) stage((it + 1) * 32, buf ^ 1);  // prefetch next tile

#pragma unroll
    for (int mt = 0; mt < 4; ++mt) {
      // A frag: lane half 0 -> K 0..7 / 16..23 ; half 1 -> K 8..15 / 24..31
      const int mrow = mw * 64 + mt * 16 + l15;
      const __bf16* ap = &As[buf][mrow * LDA + (lh ? 8 : 0)];
      const v16bf afrag = cat16(*(const v8bf*)ap, *(const v8bf*)(ap + 16));
#pragma unroll
      for (int nt = 0; nt < 2; ++nt) {
        // B frag: lane half 0 -> K 0..7 / 16..23 ; half 1 -> K 8..15 / 24..31 (N = l15)
        const int nrow = nw * 32 + nt * 16 + l15;
        const __bf16* bp = &Bs[buf][nrow * LDA + lh * 8];
        const v16bf bfrag = cat16(*(const v8bf*)bp, *(const v8bf*)(bp + 16));
        acc[mt][nt] = wmma_bf16(afrag, bfrag, acc[mt][nt]);
      }
    }
    wait_async0();        // next tile's async copies have landed in LDS
    __syncthreads();      // everyone done computing buf before it's restaged
  }

  // Epilogue: C layout VGPR v -> row (lh*8 + v), col l15.
#pragma unroll
  for (int mt = 0; mt < 4; ++mt) {
    const long mbase = m0 + mw * 64 + mt * 16 + lh * 8;
#pragma unroll
    for (int nt = 0; nt < 2; ++nt) {
      const long col = n0 + nw * 32 + nt * 16 + l15;
      const float bv = bias[col];
      if constexpr (!OPROJ) {
        __bf16* out = OutB + mbase * E_DIM + col;
#pragma unroll
        for (int v = 0; v < 8; ++v)
          out[(long)v * E_DIM] = (__bf16)(acc[mt][nt][v] + bv);
      } else {
#pragma unroll
        for (int v = 0; v < 8; ++v) {
          const long r = mbase + v;
          const long g = r / MG;
          const long orow = g * GROUPSZ + OFF + (r - g * MG);
          OutF[orow * E_DIM + col] = acc[mt][nt][v] + bv;
        }
      }
    }
  }
}

// ---------------------------------------------------------------------------
// Attention per (group, head): scores = K_gh(144x64) @ q_gh^T / 8,
// softmax over the 10 q's, attn = P(144x10) @ q_gh(10x64), written in place
// over the K slice. 4 waves; m-tiles (9) round-robin across waves.
// ---------------------------------------------------------------------------
__global__ __launch_bounds__(128)
void attn_kernel(const __bf16* __restrict__ Qb, __bf16* __restrict__ Kb)
{
  __shared__ __bf16 q_lds [16 * 64];   // [q][d]       B-frags for scores
  __shared__ __bf16 qT_lds[64 * 32];   // [d][q pad32] B-frags for attn
  __shared__ float  S_lds [144 * 16];  // scores [k][q]
  __shared__ __bf16 P_lds [144 * 32];  // probs  [k][q pad32]

  const int g    = blockIdx.x >> 4;
  const int h    = blockIdx.x & 15;
  const int tid  = threadIdx.x;
  const int lane = tid & 31;
  const int wave = tid >> 5;
  const int l15  = lane & 15;
  const int lh   = lane >> 4;

  const __bf16* qsrc  = Qb + ((long)g * NQ_) * E_DIM + h * HD;
  __bf16*       kbase = Kb + ((long)g * NKV_) * E_DIM + h * HD;

  // Load q tile (10x64, zero-padded) in both layouts.
  for (int idx = tid; idx < 16 * 64; idx += 128) {
    const int qi = idx >> 6;
    const int d  = idx & 63;
    const __bf16 val = (qi < NQ_) ? qsrc[(long)qi * E_DIM + d] : (__bf16)0.0f;
    q_lds[qi * 64 + d]  = val;
    qT_lds[d * 32 + qi] = val;
  }
  for (int idx = tid; idx < 64 * 16; idx += 128)   // zero qT pad cols 16..31
    qT_lds[(idx >> 4) * 32 + 16 + (idx & 15)] = (__bf16)0.0f;
  __syncthreads();

  // Scores: A frags straight from global (K already bf16 row-major).
  for (int mt = wave; mt < 9; mt += 4) {
    v8f acc = (v8f){0.f,0.f,0.f,0.f,0.f,0.f,0.f,0.f};
    const int m = mt * 16 + l15;
#pragma unroll
    for (int kk = 0; kk < HD; kk += 32) {
      const __bf16* ap = kbase + (long)m * E_DIM + kk + (lh ? 8 : 0);
      const v16bf afrag = cat16(*(const v8bf*)ap, *(const v8bf*)(ap + 16));
      const __bf16* bp = &q_lds[l15 * 64 + kk + lh * 8];
      const v16bf bfrag = cat16(*(const v8bf*)bp, *(const v8bf*)(bp + 16));
      acc = wmma_bf16(afrag, bfrag, acc);
    }
#pragma unroll
    for (int v = 0; v < 8; ++v)
      S_lds[(mt * 16 + lh * 8 + v) * 16 + l15] = acc[v] * 0.125f; // 1/sqrt(64)
  }
  __syncthreads();

  // Softmax over the q axis (10 values per k-row).
  for (int r = tid; r < NKV_; r += 128) {
    float vals[NQ_];
    float mx = -3.0e38f;
#pragma unroll
    for (int j = 0; j < NQ_; ++j) { vals[j] = S_lds[r * 16 + j]; mx = fmaxf(mx, vals[j]); }
    float s = 0.f;
#pragma unroll
    for (int j = 0; j < NQ_; ++j) { vals[j] = __expf(vals[j] - mx); s += vals[j]; }
    const float inv = 1.0f / s;
#pragma unroll
    for (int j = 0; j < NQ_; ++j) P_lds[r * 32 + j] = (__bf16)(vals[j] * inv);
#pragma unroll
    for (int j = NQ_; j < 32; ++j) P_lds[r * 32 + j] = (__bf16)0.0f;
  }
  __syncthreads();

  // attn = P @ q ; overwrite the K slice in place.
  for (int mt = wave; mt < 9; mt += 4) {
    const __bf16* ap = &P_lds[(mt * 16 + l15) * 32 + (lh ? 8 : 0)];
    const v16bf afrag = cat16(*(const v8bf*)ap, *(const v8bf*)(ap + 16));
#pragma unroll
    for (int nt = 0; nt < 4; ++nt) {
      const __bf16* bp = &qT_lds[(nt * 16 + l15) * 32 + lh * 8];
      const v16bf bfrag = cat16(*(const v8bf*)bp, *(const v8bf*)(bp + 16));
      v8f acc = (v8f){0.f,0.f,0.f,0.f,0.f,0.f,0.f,0.f};
      acc = wmma_bf16(afrag, bfrag, acc);
      __bf16* out = kbase + (long)(mt * 16 + lh * 8) * E_DIM + nt * 16 + l15;
#pragma unroll
      for (int v = 0; v < 8; ++v)
        out[(long)v * E_DIM] = (__bf16)acc[v];
    }
  }
}

// ---------------------------------------------------------------------------
extern "C" void kernel_launch(void* const* d_in, const int* in_sizes, int n_in,
                              void* d_out, int out_size, void* d_ws, size_t ws_size,
                              hipStream_t stream)
{
  (void)in_sizes; (void)n_in; (void)out_size; (void)ws_size;

  const float* x   = (const float*)d_in[0];
  const float* q_w = (const float*)d_in[1];
  const float* q_b = (const float*)d_in[2];
  const float* k_w = (const float*)d_in[3];
  const float* k_b = (const float*)d_in[4];
  // d_in[5]=v_w, d_in[6]=v_b are dead in the reference (v is never used).
  const float* o_w = (const float*)d_in[7];
  const float* o_b = (const float*)d_in[8];
  float* out = (float*)d_out;

  char* ws = (char*)d_ws;
  const size_t qbytes = (size_t)NGROUPS * NQ_  * E_DIM * sizeof(__bf16);
  const size_t kbytes = (size_t)NGROUPS * NKV_ * E_DIM * sizeof(__bf16);
  const size_t wbytes = (size_t)E_DIM * E_DIM * sizeof(__bf16);
  __bf16* Qbuf = (__bf16*)(ws);
  __bf16* Kbuf = (__bf16*)(ws + qbytes);
  __bf16* Wq   = (__bf16*)(ws + qbytes + kbytes);
  __bf16* Wk   = (__bf16*)(ws + qbytes + kbytes + wbytes);
  __bf16* Wo   = (__bf16*)(ws + qbytes + kbytes + 2 * wbytes);

  // 0) one-shot bf16 weight conversions (L2-resident afterwards)
  cvt_w<<<512, 256, 0, stream>>>(q_w, Wq);
  cvt_w<<<512, 256, 0, stream>>>(k_w, Wk);
  cvt_w<<<512, 256, 0, stream>>>(o_w, Wo);

  // 1) raw q_in rows -> output
  copy_qin<<<NGROUPS * NQ_, 256, 0, stream>>>(x, out);

  // 2) Q projection: 10240 x 1024, bf16 to scratch
  proj_gemm<false, NQ_, 0>
      <<<dim3(E_DIM / 128, (NGROUPS * NQ_) / 128), 256, 0, stream>>>(
          x, nullptr, Wq, q_b, Qbuf, nullptr);

  // 3) K projection: 147456 x 1024, bf16 to scratch
  proj_gemm<false, NKV_, NQ_>
      <<<dim3(E_DIM / 128, (NGROUPS * NKV_) / 128), 256, 0, stream>>>(
          x, nullptr, Wk, k_b, Kbuf, nullptr);

  // 4) per-(group,head) attention, in-place over Kbuf
  attn_kernel<<<NGROUPS * NH, 128, 0, stream>>>(Qbuf, Kbuf);

  // 5) O projection: 147456 x 1024, f32 scattered into d_out
  proj_gemm<true, NKV_, NQ_>
      <<<dim3(E_DIM / 128, (NGROUPS * NKV_) / 128), 256, 0, stream>>>(
          nullptr, Kbuf, Wo, o_b, nullptr, out);
}
